// PointNet2_56427280334962
// MI455X (gfx1250) — matrix operand order repaired and checked by reading, not compiled, not run on hardware
//
#include <hip/hip_runtime.h>
#include <hip/hip_bf16.h>
#include <hip/hip_fp16.h>

// ---------------------------------------------------------------------------
// Types for CDNA5 WMMA (wave32, 16x16x32 f16 -> f32)
// ---------------------------------------------------------------------------
typedef __attribute__((ext_vector_type(16))) _Float16 v16h;
typedef __attribute__((ext_vector_type(8)))  _Float16 v8h;
typedef __attribute__((ext_vector_type(8)))  float    v8f;

#define TILE_N 64
#define TILE_K 64

// ---------------------------------------------------------------------------
// Weight pre-pass: W (M x K f32) -> Wh (Mp x Kp f16), zero-padded.
// ---------------------------------------------------------------------------
__global__ void pad_weights_kernel(const float* __restrict__ W,
                                   _Float16* __restrict__ Wh,
                                   int M, int K, int Kp, int total)
{
    int t = blockIdx.x * blockDim.x + threadIdx.x;
    if (t >= total) return;  // total = Mp * Kp
    int r = t / Kp, c = t % Kp;
    float v = (r < M && c < K) ? W[(size_t)r * K + c] : 0.0f;
    Wh[t] = (_Float16)v;
}

// ---------------------------------------------------------------------------
// Epilogue pre-pass: fold conv bias + BN into a per-row affine (s, t),
// zero-padded to Mp so the GEMM epilogue loads are unconditional.
// mode 1: s = g*rsqrt(v+eps), t = (b-m)*s+be   (then ReLU in GEMM)
// mode 0: s = 1, t = b                         (no ReLU)
// ---------------------------------------------------------------------------
__global__ void prep_epilogue_kernel(const float* __restrict__ pb,
                                     const float* __restrict__ pg,
                                     const float* __restrict__ pbe,
                                     const float* __restrict__ pm,
                                     const float* __restrict__ pv,
                                     float* __restrict__ sArr,
                                     float* __restrict__ tArr,
                                     int M, int Mp, int mode)
{
    int r = blockIdx.x * blockDim.x + threadIdx.x;
    if (r >= Mp) return;
    float ss = 0.0f, tt = 0.0f;
    if (r < M) {
        if (mode == 1) {
            ss = pg[r] * rsqrtf(pv[r] + 1e-5f);
            tt = (pb[r] - pm[r]) * ss + pbe[r];
        } else {
            ss = 1.0f; tt = pb[r];
        }
    }
    sArr[r] = ss; tArr[r] = tt;
}

// ---------------------------------------------------------------------------
// Fused GEMM + affine epilogue (+ optional ReLU).
//   Y[b](M x Nc) = Wh(Mp x Kp, f16 zero-padded) * X[b](K x Nc)
//   y = acc * s[row] + t[row];  relu? max(y,0)
// One 16x16 WMMA tile per wave; 4 waves/block cover a 16x64 output tile;
// two v_wmma per K-stage (TILE_K=64).
//  - A fragments: direct, unconditional 16B global loads from padded Wh
//    (exact ISA 16-bit A lane striping; zeros in the padding).
//  - B tile: two-phase staging -- 8 clamped float4 loads into registers
//    (single wait), then f16 convert + transposed LDS stores; fragments
//    read back as contiguous b128 ds loads.
// Requires Nc % 64 == 0 (true at every call site).
// ---------------------------------------------------------------------------
__global__ __launch_bounds__(128) void gemm_bn_relu_kernel(
    const _Float16* __restrict__ Wh, const float* __restrict__ X,
    float* __restrict__ Y,
    const float* __restrict__ sArr, const float* __restrict__ tArr,
    int M, int K, int Kp, int Nc, int relu)
{
    __shared__ __align__(16) _Float16 Bs[TILE_N * TILE_K];   // [col][k] (transposed)

    const int tid   = threadIdx.x;
    const int wave  = tid >> 5;
    const int lane  = tid & 31;
    const int bb    = blockIdx.z;
    const int tileM = blockIdx.y * 16;
    const int tileN = blockIdx.x * TILE_N;

    const float* Xb = X + (size_t)bb * (size_t)K * (size_t)Nc;

    v8f acc = {};

    const int m16  = lane & 15;
    const int kbA  = (lane < 16) ? 0 : 8;    // A fragment K sub-base (halves)
    const int kbB  = (lane < 16) ? 0 : 8;    // B fragment K base
    const int scol = wave * 16 + m16;        // this wave's column within tile

    const _Float16* WhRow = Wh + (size_t)(tileM + m16) * Kp;

    for (int kk = 0; kk < Kp; kk += TILE_K) {
        // ---- phase 1: 8 clamped float4 loads into registers (one wait) ----
        float4 tv[8];
        #pragma unroll
        for (int i = 0; i < 8; ++i) {
            int id  = tid + i * 128;         // 0..1023
            int r   = id >> 4;               // k row 0..63
            int c4  = (id & 15) * 4;         // col group 0,4,...,60
            int gk  = kk + r;
            int gkc = gk < K ? gk : K - 1;   // clamp only; zero-A padding kills it
            tv[i] = *(const float4*)(Xb + (size_t)gkc * Nc + tileN + c4);
        }
        // ---- phase 2: convert + transposed LDS stores ----
        #pragma unroll
        for (int i = 0; i < 8; ++i) {
            int id = tid + i * 128;
            int r  = id >> 4;
            int c4 = (id & 15) * 4;
            Bs[(c4 + 0) * TILE_K + r] = (_Float16)tv[i].x;
            Bs[(c4 + 1) * TILE_K + r] = (_Float16)tv[i].y;
            Bs[(c4 + 2) * TILE_K + r] = (_Float16)tv[i].z;
            Bs[(c4 + 3) * TILE_K + r] = (_Float16)tv[i].w;
        }
        __syncthreads();

        // ---- 2 WMMAs per stage ----
        #pragma unroll
        for (int j = 0; j < 2; ++j) {
            // A fragment: direct 16B global loads from padded f16 weights
            const v8h a_lo = *(const v8h*)(WhRow + kk + j * 32 + kbA);
            const v8h a_hi = *(const v8h*)(WhRow + kk + j * 32 + 16 + kbA);
            v16h af = __builtin_shufflevector(a_lo, a_hi,
                0,1,2,3,4,5,6,7,8,9,10,11,12,13,14,15);

            const v8h b_lo = *(const v8h*)&Bs[scol * TILE_K + j * 32 + kbB];
            const v8h b_hi = *(const v8h*)&Bs[scol * TILE_K + j * 32 + kbB + 16];
            v16h bf = __builtin_shufflevector(b_lo, b_hi,
                0,1,2,3,4,5,6,7,8,9,10,11,12,13,14,15);

            acc = __builtin_amdgcn_wmma_f32_16x16x32_f16(
                false, af, false, bf, (short)0, acc, false, false);
        }
        __syncthreads();
    }

    // --- epilogue: D layout -> row = tileM + j + (lane<16?0:8), col = lane&15 ---
    const int n = tileN + scol;
    const int mbase = tileM + ((lane >> 4) << 3);
    float* Yb = Y + (size_t)bb * (size_t)M * (size_t)Nc;
    #pragma unroll
    for (int j = 0; j < 8; ++j) {
        int row = mbase + j;                 // always < Mp: s/t loads unconditional
        float v = fmaf(acc[j], sArr[row], tArr[row]);
        if (relu) v = fmaxf(v, 0.0f);
        if (row < M) Yb[(size_t)row * Nc + n] = v;
    }
}

// ---------------------------------------------------------------------------
// Farthest point sampling: one block per batch, dist in registers,
// block argmax tree in LDS. Records index BEFORE update (matches lax.scan).
// ---------------------------------------------------------------------------
#define FPS_T 256
#define FPS_MAXP 64

__global__ __launch_bounds__(FPS_T) void fps_kernel(
    const float* __restrict__ xyzT, int N, int npoint, int* __restrict__ out)
{
    __shared__ float rmax[FPS_T];
    __shared__ int   rarg[FPS_T];
    __shared__ int   curIdx;
    const int b = blockIdx.x, tid = threadIdx.x;
    const float* P = xyzT + (size_t)b * N * 3;

    float dist[FPS_MAXP];
    #pragma unroll
    for (int j = 0; j < FPS_MAXP; ++j) dist[j] = 1e10f;
    if (tid == 0) curIdx = 0;
    __syncthreads();

    for (int it = 0; it < npoint; ++it) {
        int far = curIdx;
        if (tid == 0) out[(size_t)b * npoint + it] = far;
        float cx = P[far * 3 + 0], cy = P[far * 3 + 1], cz = P[far * 3 + 2];
        float best = -1.0f; int bestI = 0;
        #pragma unroll
        for (int j = 0; j < FPS_MAXP; ++j) {
            int i = tid + j * FPS_T;
            if (i >= N) break;
            float dx = P[i * 3 + 0] - cx;
            float dy = P[i * 3 + 1] - cy;
            float dz = P[i * 3 + 2] - cz;
            float d  = dx * dx + dy * dy + dz * dz;
            float dd = fminf(dist[j], d);
            dist[j] = dd;
            if (dd > best) { best = dd; bestI = i; }
        }
        rmax[tid] = best; rarg[tid] = bestI;
        __syncthreads();
        for (int s = FPS_T / 2; s > 0; s >>= 1) {
            if (tid < s && rmax[tid + s] > rmax[tid]) {
                rmax[tid] = rmax[tid + s]; rarg[tid] = rarg[tid + s];
            }
            __syncthreads();
        }
        if (tid == 0) curIdx = rarg[0];
        __syncthreads();
    }
}

// (B,3,N) -> (B,N,3)
__global__ void to_points_kernel(const float* __restrict__ src,
                                 float* __restrict__ dst, int N, int total)
{
    int t = blockIdx.x * blockDim.x + threadIdx.x;
    if (t >= total) return;
    int b = t / N, n = t % N;
    #pragma unroll
    for (int c = 0; c < 3; ++c)
        dst[((size_t)b * N + n) * 3 + c] = src[((size_t)b * 3 + c) * N + n];
}

// gather sampled centers: (B,N,3) + idx(B,S) -> (B,S,3)
__global__ void gather_xyz_kernel(const float* __restrict__ xyzT,
                                  const int* __restrict__ idx,
                                  float* __restrict__ dst, int N, int S, int total)
{
    int t = blockIdx.x * blockDim.x + threadIdx.x;
    if (t >= total) return;
    int b = t / S, s = t % S;
    int i = idx[(size_t)b * S + s];
    #pragma unroll
    for (int c = 0; c < 3; ++c)
        dst[((size_t)b * S + s) * 3 + c] = xyzT[((size_t)b * N + i) * 3 + c];
}

// ball query: first K indices (scan order == smallest indices) with d <= r^2
__global__ void ball_query_kernel(const float* __restrict__ xyzT,
                                  const float* __restrict__ newT,
                                  int N, int S, float r2, int K,
                                  int* __restrict__ gi, int total)
{
    int t = blockIdx.x * blockDim.x + threadIdx.x;
    if (t >= total) return;
    int b = t / S, s = t % S;
    const float* P = xyzT + (size_t)b * N * 3;
    const float* q = newT + ((size_t)b * S + s) * 3;
    int* out = gi + ((size_t)b * S + s) * K;
    int cnt = 0, first = 0;
    for (int n = 0; n < N && cnt < K; ++n) {
        float dx = P[n * 3 + 0] - q[0];
        float dy = P[n * 3 + 1] - q[1];
        float dz = P[n * 3 + 2] - q[2];
        float d = dx * dx + dy * dy + dz * dz;
        if (d <= r2) { if (cnt == 0) first = n; out[cnt++] = n; }
    }
    for (; cnt < K; ++cnt) out[cnt] = first;
}

// build grouped GEMM operand G (B, C+3, K, S): features then relative xyz
__global__ void group_kernel(const float* __restrict__ pts,   // (B,C,Np)
                             const float* __restrict__ xyzT,  // (B,Np,3)
                             const float* __restrict__ newT,  // (B,S,3)
                             const int* __restrict__ gi,      // (B,S,K)
                             float* __restrict__ G,
                             int C, int Np, int S, int K, int total)
{
    int t = blockIdx.x * blockDim.x + threadIdx.x;
    if (t >= total) return;  // total = B*K*S
    int s = t % S, k = (t / S) % K, b = t / (S * K);
    int i = gi[((size_t)b * S + s) * K + k];
    size_t KS = (size_t)K * S;
    size_t gbase = (size_t)b * (C + 3) * KS + (size_t)k * S + s;
    const float* pbm = pts + (size_t)b * C * Np;
    for (int c = 0; c < C; ++c) G[gbase + (size_t)c * KS] = pbm[(size_t)c * Np + i];
    const float* xp = xyzT + ((size_t)b * Np + i) * 3;
    const float* q  = newT + ((size_t)b * S + s) * 3;
    G[gbase + (size_t)(C + 0) * KS] = xp[0] - q[0];
    G[gbase + (size_t)(C + 1) * KS] = xp[1] - q[1];
    G[gbase + (size_t)(C + 2) * KS] = xp[2] - q[2];
}

// max over K: (B,C,K,S) -> slab of (B,Ctot,S) at channel offset
__global__ void maxpool_kernel(const float* __restrict__ G,
                               float* __restrict__ out,
                               int C, int K, int S, int Ctot, int chanOff, int total)
{
    int t = blockIdx.x * blockDim.x + threadIdx.x;
    if (t >= total) return;  // total = B*C*S
    int s = t % S, c = (t / S) % C, b = t / (S * C);
    const float* g = G + ((size_t)b * C + c) * (size_t)K * S + s;
    float m = g[0];
    for (int k = 1; k < K; ++k) m = fmaxf(m, g[(size_t)k * S]);
    out[((size_t)b * Ctot + chanOff + c) * S + s] = m;
}

// 3-NN with inverse-distance weights (normalized)
__global__ void knn3_kernel(const float* __restrict__ x1T, const float* __restrict__ x2T,
                            int N, int S, int* __restrict__ idx3, float* __restrict__ w3,
                            int total)
{
    int t = blockIdx.x * blockDim.x + threadIdx.x;
    if (t >= total) return;  // total = B*N
    int b = t / N, n = t % N;
    const float* q = x1T + ((size_t)b * N + n) * 3;
    const float* P = x2T + (size_t)b * S * 3;
    float d0 = 1e30f, d1 = 1e30f, d2 = 1e30f; int i0 = 0, i1 = 0, i2 = 0;
    for (int s = 0; s < S; ++s) {
        float dx = P[s * 3 + 0] - q[0];
        float dy = P[s * 3 + 1] - q[1];
        float dz = P[s * 3 + 2] - q[2];
        float d = dx * dx + dy * dy + dz * dz;
        if (d < d0)      { d2 = d1; i2 = i1; d1 = d0; i1 = i0; d0 = d; i0 = s; }
        else if (d < d1) { d2 = d1; i2 = i1; d1 = d;  i1 = s; }
        else if (d < d2) { d2 = d;  i2 = s; }
    }
    float w0 = 1.0f / fmaxf(d0, 1e-10f);
    float w1 = 1.0f / fmaxf(d1, 1e-10f);
    float w2 = 1.0f / fmaxf(d2, 1e-10f);
    float inv = 1.0f / (w0 + w1 + w2);
    size_t base = ((size_t)b * N + n) * 3;
    idx3[base + 0] = i0; idx3[base + 1] = i1; idx3[base + 2] = i2;
    w3[base + 0] = w0 * inv; w3[base + 1] = w1 * inv; w3[base + 2] = w2 * inv;
}

// weighted gather: dst slab (B,Ctot,N) at chanOff from p2 (B,C2,S)
__global__ void interp_kernel(const float* __restrict__ p2,
                              const int* __restrict__ idx3, const float* __restrict__ w3,
                              float* __restrict__ dst,
                              int C2, int S, int N, int Ctot, int chanOff, int total)
{
    int t = blockIdx.x * blockDim.x + threadIdx.x;
    if (t >= total) return;  // total = B*C2*N
    int n = t % N, c = (t / N) % C2, b = t / (N * C2);
    size_t base = ((size_t)b * N + n) * 3;
    const float* p = p2 + ((size_t)b * C2 + c) * S;
    float v = w3[base + 0] * p[idx3[base + 0]]
            + w3[base + 1] * p[idx3[base + 1]]
            + w3[base + 2] * p[idx3[base + 2]];
    dst[((size_t)b * Ctot + chanOff + c) * N + n] = v;
}

// channel-slab copy for skip connections
__global__ void copychan_kernel(const float* __restrict__ src, float* __restrict__ dst,
                                int C, int N, int Ctot, int chanOff, int total)
{
    int t = blockIdx.x * blockDim.x + threadIdx.x;
    if (t >= total) return;  // total = B*C*N
    int n = t % N, c = (t / N) % C, b = t / (N * C);
    dst[((size_t)b * Ctot + chanOff + c) * N + n] = src[((size_t)b * C + c) * N + n];
}

// ---------------------------------------------------------------------------
// Host orchestration
// ---------------------------------------------------------------------------
struct CB { const float *b, *be, *g, *m, *v, *w; };

static inline CB readCB(void* const* din, int& i) {
    CB p;
    p.b  = (const float*)din[i++];  p.be = (const float*)din[i++];
    p.g  = (const float*)din[i++];  p.m  = (const float*)din[i++];
    p.v  = (const float*)din[i++];  p.w  = (const float*)din[i++];
    return p;
}
static inline CB readCV(void* const* din, int& i) {
    CB p;
    p.b = (const float*)din[i++];   p.w = (const float*)din[i++];
    p.be = p.g = p.m = p.v = p.b;   // unused in mode 0
    return p;
}

extern "C" void kernel_launch(void* const* d_in, const int* in_sizes, int n_in,
                              void* d_out, int out_size, void* d_ws, size_t ws_size,
                              hipStream_t stream)
{
    (void)in_sizes; (void)n_in; (void)out_size; (void)ws_size;
    const int Bb = 2, N0 = 16384;
    const float* xyz = (const float*)d_in[0];

    // ---- parameter unpack (jax pytree order: dict keys sorted) ----
    int pi = 1;
    const int fpn[4] = {2, 2, 2, 3};
    CB fpcb[4][3];
    for (int i = 0; i < 4; ++i)
        for (int j = 0; j < fpn[i]; ++j) fpcb[i][j] = readCB(d_in, pi);
    // head keys sorted: conv1,f1,f2,fea,o1,o2,off,s1,s2,sem
    CB hconv1 = readCB(d_in, pi);
    CB hf1 = readCB(d_in, pi), hf2 = readCB(d_in, pi);
    CB hfea = readCV(d_in, pi);
    CB ho1 = readCB(d_in, pi), ho2 = readCB(d_in, pi);
    CB hoff = readCV(d_in, pi);
    CB hs1 = readCB(d_in, pi), hs2 = readCB(d_in, pi);
    CB hsem = readCV(d_in, pi);
    CB sacb[4][2][3];
    for (int l = 0; l < 4; ++l)
        for (int br = 0; br < 2; ++br)
            for (int j = 0; j < 3; ++j) sacb[l][br][j] = readCB(d_in, pi);

    // ---- workspace bump allocator (16B-aligned slabs for vector loads) ----
    float* wsf = (float*)d_ws;
    size_t woff = 0;
    auto alloc = [&](size_t nf) {
        nf = (nf + 3) & ~(size_t)3;
        float* p = wsf + woff; woff += nf; return p;
    };

    const int lvlN[5] = {16384, 1024, 256, 64, 16};
    const int pchan[5] = {3, 96, 256, 512, 1024};

    float* xyzT[5];
    for (int l = 0; l < 5; ++l) xyzT[l] = alloc((size_t)Bb * lvlN[l] * 3);
    int*   fpsidx = (int*)alloc((size_t)Bb * 1024);
    int*   gi     = (int*)alloc((size_t)Bb * 1024 * 32);
    int*   knni   = (int*)alloc((size_t)Bb * N0 * 3);
    float* knnw   = alloc((size_t)Bb * N0 * 3);
    _Float16* Wh  = (_Float16*)alloc(200704);  // 401408 halfs >= max Mp*Kp (393216)
    float* stS    = alloc(512);                // per-row epilogue scale (Mp <= 512)
    float* stT    = alloc(512);                // per-row epilogue shift
    float* pbuf[5];
    pbuf[0] = (float*)xyz;  // ps[0] = xyz as (B,3,N)
    for (int l = 1; l < 5; ++l) pbuf[l] = alloc((size_t)Bb * pchan[l] * lvlN[l]);
    float* f3 = alloc((size_t)Bb * 256 * 64);
    float* f2 = alloc((size_t)Bb * 256 * 256);
    float* f1 = alloc((size_t)Bb * 128 * 1024);
    float* f0 = alloc((size_t)Bb * 128 * N0);
    float* headx = alloc((size_t)Bb * 128 * N0);
    const size_t BUF = 4u * 1024u * 1024u + 65536u;
    float* bufA = alloc(BUF);
    float* bufB = alloc(BUF);

    auto blocks = [](int total) { return dim3((total + 255) / 256); };

    auto gemm = [&](const CB& p, const float* X, float* Y, int M, int K, int Nc, int mode) {
        const int Mp = (M + 15) & ~15;
        const int Kp = (K + 63) & ~63;
        const int tot = Mp * Kp;
        pad_weights_kernel<<<blocks(tot), 256, 0, stream>>>(p.w, Wh, M, K, Kp, tot);
        prep_epilogue_kernel<<<blocks(Mp), 256, 0, stream>>>(
            p.b, p.g, p.be, p.m, p.v, stS, stT, M, Mp, mode);
        dim3 g(Nc / TILE_N, Mp / 16, Bb);
        gemm_bn_relu_kernel<<<g, dim3(128), 0, stream>>>(
            Wh, X, Y, stS, stT, M, K, Kp, Nc, mode == 1 ? 1 : 0);
    };

    // ---- xyzT for level 0 ----
    to_points_kernel<<<blocks(Bb * N0), 256, 0, stream>>>(xyz, xyzT[0], N0, Bb * N0);

    // ---- Set Abstraction (MSG) levels ----
    const float sa_r[4][2] = {{0.05f, 0.1f}, {0.1f, 0.2f}, {0.2f, 0.4f}, {0.4f, 0.8f}};
    const int   sa_K[2]    = {16, 32};
    const int   sa_mlp[4][2][3] = {
        {{16, 16, 32},   {32, 32, 64}},
        {{64, 64, 128},  {64, 64, 128}},
        {{128, 196, 256},{128, 196, 256}},
        {{256, 256, 512},{256, 256, 512}}};

    for (int l = 0; l < 4; ++l) {
        const int Np = lvlN[l], S = lvlN[l + 1], Cprev = pchan[l];
        fps_kernel<<<dim3(Bb), FPS_T, 0, stream>>>(xyzT[l], Np, S, fpsidx);
        gather_xyz_kernel<<<blocks(Bb * S), 256, 0, stream>>>(
            xyzT[l], fpsidx, xyzT[l + 1], Np, S, Bb * S);
        int chanOff = 0;
        for (int br = 0; br < 2; ++br) {
            const int K = sa_K[br];
            const float r = sa_r[l][br];
            ball_query_kernel<<<blocks(Bb * S), 256, 0, stream>>>(
                xyzT[l], xyzT[l + 1], Np, S, r * r, K, gi, Bb * S);
            group_kernel<<<blocks(Bb * K * S), 256, 0, stream>>>(
                pbuf[l], xyzT[l], xyzT[l + 1], gi, bufA, Cprev, Np, S, K, Bb * K * S);
            const int Ncols = K * S;
            int cin = Cprev + 3;
            const float* x = bufA;
            for (int j = 0; j < 3; ++j) {
                float* y = (x == bufA) ? bufB : bufA;
                gemm(sacb[l][br][j], x, y, sa_mlp[l][br][j], cin, Ncols, 1);
                cin = sa_mlp[l][br][j]; x = y;
            }
            maxpool_kernel<<<blocks(Bb * cin * S), 256, 0, stream>>>(
                x, pbuf[l + 1], cin, K, S, pchan[l + 1], chanOff, Bb * cin * S);
            chanOff += cin;
        }
    }

    // ---- Feature Propagation levels ----
    struct FPL { int la, lb; const float* p1; int C1; const float* p2; int C2;
                 int nl; float* out; };
    FPL fpl[4] = {
        {3, 4, pbuf[3], 512, pbuf[4], 1024, 2, f3},
        {2, 3, pbuf[2], 256, f3,      256,  2, f2},
        {1, 2, pbuf[1], 96,  f2,      256,  2, f1},
        {0, 1, nullptr, 0,   f1,      128,  3, f0}};
    const int fp_out[4][3] = {{256, 256, 0}, {256, 256, 0}, {256, 128, 0}, {128, 128, 128}};

    for (int i = 0; i < 4; ++i) {
        const int N = lvlN[fpl[i].la], S = lvlN[fpl[i].lb];
        knn3_kernel<<<blocks(Bb * N), 256, 0, stream>>>(
            xyzT[fpl[i].la], xyzT[fpl[i].lb], N, S, knni, knnw, Bb * N);
        const int Ctot = fpl[i].C1 + fpl[i].C2;
        if (fpl[i].p1)
            copychan_kernel<<<blocks(Bb * fpl[i].C1 * N), 256, 0, stream>>>(
                fpl[i].p1, bufA, fpl[i].C1, N, Ctot, 0, Bb * fpl[i].C1 * N);
        interp_kernel<<<blocks(Bb * fpl[i].C2 * N), 256, 0, stream>>>(
            fpl[i].p2, knni, knnw, bufA, fpl[i].C2, S, N, Ctot, fpl[i].C1,
            Bb * fpl[i].C2 * N);
        int cin = Ctot;
        const float* x = bufA;
        for (int j = 0; j < fpl[i].nl; ++j) {
            float* y = (j == fpl[i].nl - 1) ? fpl[i].out
                                            : ((x == bufA) ? bufB : bufA);
            gemm(fpcb[i][j], x, y, fp_out[i][j], cin, N, 1);
            cin = fp_out[i][j]; x = y;
        }
    }

    // ---- Head ----
    float* outp = (float*)d_out;
    float* out_off = outp;
    float* out_sem = outp + (size_t)Bb * 3 * N0;
    float* out_fea = outp + (size_t)Bb * 5 * N0;

    gemm(hconv1, f0, headx, 128, 128, N0, 1);
    // offset branch
    gemm(ho1, headx, bufA, 64, 128, N0, 1);
    gemm(ho2, bufA, bufB, 32, 64, N0, 1);
    gemm(hoff, bufB, out_off, 3, 32, N0, 0);
    // semantic branch
    gemm(hs1, headx, bufA, 64, 128, N0, 1);
    gemm(hs2, bufA, bufB, 32, 64, N0, 1);
    gemm(hsem, bufB, out_sem, 2, 32, N0, 0);
    // feature branch
    gemm(hf1, headx, bufA, 64, 128, N0, 1);
    gemm(hf2, bufA, bufB, 64, 64, N0, 1);
    gemm(hfea, bufB, out_fea, 64, 64, N0, 0);
}
